// AttentionEntropyLoss_34986803593382
// MI455X (gfx1250) — matrix operand order repaired and checked by reading, not compiled, not run on hardware
//
#include <hip/hip_runtime.h>
#include <hip/hip_bf16.h>

typedef __attribute__((ext_vector_type(16))) _Float16 v16h;
typedef __attribute__((ext_vector_type(8)))  _Float16 v8h;
typedef __attribute__((ext_vector_type(8)))  float    v8f;

#define S_LEN      2048
#define D_HEAD_C   64
#define KB         64      // keys staged in LDS per block
#define QTILE      128     // queries per workgroup (8 waves x 16)
#define LDS_STRIDE 72      // f16 elems per K row: 64 + 8 pad (kills bank conflicts)

// Q pre-scale: (1/sqrt(64)) * log2(e)  -> scores come out in log2 units
#define QSCALE 0.18033688011112042f
#define LN2f   0.69314718055994531f

// Workgroup split-barrier with DS-only visibility: unlike __syncthreads()'
// full acq-rel fence, this does NOT drain LOADcnt, so the register-staged
// global loads for future K blocks keep flying across the barrier.
__device__ __forceinline__ void wg_barrier_ds() {
    asm volatile(
        "s_wait_dscnt 0x0\n\t"
        "s_barrier_signal -1\n\t"
        "s_barrier_wait -1"
        ::: "memory");
}

// ws[0] = sum(entropy), ws[1] = sum(entropy^2)
__global__ void attn_entropy_init(float* ws) {
    ws[0] = 0.0f;
    ws[1] = 0.0f;
}

__global__ __launch_bounds__(256) void attn_entropy_main(
    const float* __restrict__ Q, const float* __restrict__ K,
    float* __restrict__ ws)
{
    __shared__ __align__(16) _Float16 kbuf[KB * LDS_STRIDE];

    const int bh    = blockIdx.y;          // fused (b,h): 0..31
    const int qtile = blockIdx.x;          // 0..15 (128 queries each)
    const int tid   = threadIdx.x;
    const int lane  = tid & 31;
    const int wave  = tid >> 5;
    const int l16   = lane & 15;
    const int hi    = lane >> 4;           // lane half selector

    const float* Qb = Q + (size_t)bh * S_LEN * D_HEAD_C;
    const float* Kb = K + (size_t)bh * S_LEN * D_HEAD_C;

    // ---- Build this lane's WMMA B fragments: B = Q^T (d x 16 queries) ----
    // B frag, contiguous k order (same as A): element i -> K-row (d) = base + i + 16*hi,
    // column (query) = lane%16.  Scale folds in 1/sqrt(D) and log2(e).
    const int    qrow = qtile * QTILE + wave * 16 + l16;
    const float* qp   = Qb + (size_t)qrow * D_HEAD_C;
    v16h bq0, bq1;
    {
        const int c0 = 16 * hi;          // frag0 covers d = c0 .. c0+15
        const int c1 = 32 + 16 * hi;     // frag1 covers d = c1 .. c1+15
        #pragma unroll
        for (int i = 0; i < 16; ++i) {
            bq0[i] = (_Float16)(qp[c0 + i] * QSCALE);
            bq1[i] = (_Float16)(qp[c1 + i] * QSCALE);
        }
    }

    // Max-free accumulation (inputs bounded: |s_log2| < ~10, fp32 exp2 safe):
    //   Z = sum 2^s,  S1 = sum 2^s * s   (s in log2 units)
    // 4 independent accumulator pairs; merged once at the very end.
    float Zp[4] = {0.f, 0.f, 0.f, 0.f};
    float Sp[4] = {0.f, 0.f, 0.f, 0.f};

    // ---- distance-2 software pipeline: regs -> (convert) -> LDS -> WMMA --
    const int row = tid >> 2;              // 0..63 : K row inside block
    const int qu  = (tid & 3) * 16;        // quarter of the 64-wide row
    const float* kbase = Kb + (size_t)row * D_HEAD_C + qu;

    float4 c0v, c1v, c2v, c3v;             // block kb   (converted this iter)
    float4 n0v, n1v, n2v, n3v;             // block kb+1 (staged)
    {
        const float4* p0 = (const float4*)kbase;
        c0v = p0[0]; c1v = p0[1]; c2v = p0[2]; c3v = p0[3];
        const float4* p1 = (const float4*)(kbase + (size_t)KB * D_HEAD_C);
        n0v = p1[0]; n1v = p1[1]; n2v = p1[2]; n3v = p1[3];
    }

    const int nblk = S_LEN / KB;
    for (int kb = 0; kb < nblk; ++kb) {
        // -------- convert staged regs (block kb) to f16, store to LDS -----
        {
            float f[16] = { c0v.x, c0v.y, c0v.z, c0v.w,
                            c1v.x, c1v.y, c1v.z, c1v.w,
                            c2v.x, c2v.y, c2v.z, c2v.w,
                            c3v.x, c3v.y, c3v.z, c3v.w };
            v8h lo8, hi8;
            #pragma unroll
            for (int i = 0; i < 8; ++i) {
                lo8[i] = (_Float16)f[i];
                hi8[i] = (_Float16)f[8 + i];
            }
            v8h* dst = (v8h*)&kbuf[row * LDS_STRIDE + qu];
            dst[0] = lo8;
            dst[1] = hi8;
        }
        wg_barrier_ds();

        // issue block kb+2's global loads now: they get ~2 compute blocks
        // of latency cover and are NOT drained by the DS-only barriers
        float4 g0 = n0v, g1 = n1v, g2 = n2v, g3 = n3v;
        if (kb + 2 < nblk) {
            const float4* p =
                (const float4*)(kbase + (size_t)(kb + 2) * KB * D_HEAD_C);
            g0 = p[0]; g1 = p[1]; g2 = p[2]; g3 = p[3];
        }
        if (kb + 3 < nblk) {   // L2 prefetch at distance 3
            __builtin_prefetch(kbase + (size_t)(kb + 3) * KB * D_HEAD_C, 0, 3);
        }

        // -------- 4 sub-tiles of 16 keys: D[key,query] = K * Q^T ----------
        v8f acc[4];
        #pragma unroll
        for (int t = 0; t < 4; ++t) {
            // A frag, contiguous k order (same as B): elems 0-15 -> d = 16*hi + 0..15
            const _Float16* rp = &kbuf[(t * 16 + l16) * LDS_STRIDE];
            v8h a00 = *(const v8h*)(rp +      16 * hi);
            v8h a01 = *(const v8h*)(rp +  8 + 16 * hi);
            v8h a10 = *(const v8h*)(rp + 32 + 16 * hi);
            v8h a11 = *(const v8h*)(rp + 40 + 16 * hi);
            v16h a0 = __builtin_shufflevector(a00, a01,
                        0,1,2,3,4,5,6,7,8,9,10,11,12,13,14,15);
            v16h a1 = __builtin_shufflevector(a10, a11,
                        0,1,2,3,4,5,6,7,8,9,10,11,12,13,14,15);
            v8f c = {};
            c = __builtin_amdgcn_wmma_f32_16x16x32_f16(false, a0, false, bq0,
                                                       (short)0, c, false, false);
            c = __builtin_amdgcn_wmma_f32_16x16x32_f16(false, a1, false, bq1,
                                                       (short)0, c, false, false);
            acc[t] = c;
        }

        // -------- entropy partials: 1 exp2 + 1 add + 1 fma per score ------
        #pragma unroll
        for (int t = 0; t < 4; ++t) {
            #pragma unroll
            for (int r = 0; r < 8; ++r) {
                float s = acc[t][r];                       // log2-domain score
                float e = __builtin_amdgcn_exp2f(s);       // v_exp_f32
                Zp[t] += e;
                Sp[t] = fmaf(e, s, Sp[t]);
            }
        }

        wg_barrier_ds();    // fragment ds_loads done; LDS safe to overwrite

        c0v = n0v; c1v = n1v; c2v = n2v; c3v = n3v;   // rotate pipeline
        n0v = g0;  n1v = g1;  n2v = g2;  n3v = g3;
    }

    // -------- finalize: merge accumulators, entropy per query -------------
    float Zs = (Zp[0] + Zp[1]) + (Zp[2] + Zp[3]);
    float S1 = (Sp[0] + Sp[1]) + (Sp[2] + Sp[3]);
    Zs += __shfl_xor(Zs, 16, 32);     // merge key-halves (lanes q / q+16)
    S1 += __shfl_xor(S1, 16, 32);

    // entropy(nats) = ln2 * (log2(Z) - S1/Z)
    float ent = LN2f * (__builtin_amdgcn_logf(Zs) - S1 / Zs);   // v_log_f32
    float e1 = ent, e2 = ent * ent;
    #pragma unroll
    for (int off = 8; off >= 1; off >>= 1) {   // sum lanes 0..15 (16 queries)
        e1 += __shfl_xor(e1, off, 32);
        e2 += __shfl_xor(e2, off, 32);
    }
    if (lane == 0) {
        atomicAdd(&ws[0], e1);
        atomicAdd(&ws[1], e2);
    }
}

__global__ void attn_entropy_final(const float* __restrict__ ws,
                                   float* __restrict__ out)
{
    const float N  = 2.0f * 16.0f * 2048.0f;   // B*H*S entropies
    const float s  = ws[0];
    const float sq = ws[1];
    const float mean = s / N;
    const float loss = (mean - 5.0f) * (mean - 5.0f);
    const float var  = (sq - s * s / N) / (N - 1.0f);  // ddof=1
    out[0] = loss;
    out[1] = mean;
    out[2] = sqrtf(fmaxf(var, 0.0f));
}

extern "C" void kernel_launch(void* const* d_in, const int* in_sizes, int n_in,
                              void* d_out, int out_size, void* d_ws, size_t ws_size,
                              hipStream_t stream) {
    const float* Q = (const float*)d_in[0];
    const float* K = (const float*)d_in[1];
    // d_in[2..4] (values, k_cg, v_cg) are unused by the reference.
    float* ws  = (float*)d_ws;
    float* out = (float*)d_out;

    attn_entropy_init<<<dim3(1), dim3(1), 0, stream>>>(ws);

    dim3 grid(S_LEN / QTILE, 2 * 16);   // 16 query tiles x 32 (b,h)
    attn_entropy_main<<<grid, dim3(256), 0, stream>>>(Q, K, ws);

    attn_entropy_final<<<dim3(1), dim3(1), 0, stream>>>(ws, out);
}
